// GPT_42571715838398
// MI455X (gfx1250) — matrix operand order repaired and checked by reading, not compiled, not run on hardware
//
#include <hip/hip_runtime.h>
#include <hip/hip_bf16.h>

// ---------------- model constants ----------------
#define LNUM 6
#define HNUM 8
#define CDIM 512
#define DLOW 64
#define DHEAD 64
#define VOCAB 50304
#define BSZ 2
#define TSEQ 1024
#define ENUM 8
#define NSHARED 2
#define NROUTED 6
#define TOPK 2
#define HIDDEN 1024
#define NTOK (BSZ * TSEQ)

// ---------------- WMMA GEMM ----------------
typedef __attribute__((ext_vector_type(16))) __bf16 v16bf;
typedef __attribute__((ext_vector_type(8)))  __bf16 v8bf;
typedef __attribute__((ext_vector_type(8)))  float  v8f;

#define TM 128
#define TN 128
#define TKK 32
#define LDSP 48   // padded LDS row stride in bf16 elements (96B, 16B-aligned)

__device__ __forceinline__ v8bf pack8(const float* f)
{
    v8bf r;
#pragma unroll
    for (int j = 0; j < 8; j++) r[j] = (__bf16)f[j];
    return r;
}

// Assemble a 16-element bf16 fragment from two 16B-aligned LDS runs.
__device__ __forceinline__ v16bf frag16(const __bf16* p0, const __bf16* p1)
{
    v8bf lo = *(const v8bf*)p0;
    v8bf hi = *(const v8bf*)p1;
    return __builtin_shufflevector(lo, hi, 0, 1, 2, 3, 4, 5, 6, 7,
                                   8, 9, 10, 11, 12, 13, 14, 15);
}

// C = alpha*(A@B) + bias + (accum ? C : 0); f32 in/out, bf16 WMMA core.
// REQUIRES: M % TM == 0 and K % TKK == 0 (true for every call below).
// 2-level batch: blockIdx.z = b1*nb2 + b2. transB: B(k,n) read from Bm[n*ldb+k].
// causal==1: skip fully-masked output blocks (j > i everywhere).
// causal==2: truncate K loop at blockM+TM (A columns beyond that are zero).
// Double-buffered LDS pipeline: one barrier per K-step, global loads for the
// next tile in flight during the current tile's 8 WMMAs.
__global__ __launch_bounds__(256) void gemm_bf16_wmma(
    const float* __restrict__ A, int lda, long long sA1, long long sA2,
    const float* __restrict__ Bm, int ldb, long long sB1, long long sB2,
    float* __restrict__ Cm, int ldc, long long sC1, long long sC2,
    const float* __restrict__ bias,
    int M, int N, int K, int nb2, float alpha, int transB, int accum, int causal)
{
    __shared__ __bf16 As[2][TM][LDSP];   // (m, k) row-major
    __shared__ __bf16 BsT[2][TN][LDSP];  // (n, k): B transposed so frags are contiguous

    const int blockM = blockIdx.y * TM, blockN = blockIdx.x * TN;
    if (causal == 1 && blockN >= blockM + TM) return;   // fully masked tile

    int b1 = blockIdx.z / nb2, b2 = blockIdx.z % nb2;
    A  += (long long)b1 * sA1 + (long long)b2 * sA2;
    Bm += (long long)b1 * sB1 + (long long)b2 * sB2;
    Cm += (long long)b1 * sC1 + (long long)b2 * sC2;

    const int tid  = threadIdx.x;
    const int lane = tid & 31, wave = tid >> 5;
    const int mw = (wave & 3) * 32;      // wave's 32-row strip
    const int nw = (wave >> 2) * 64;     // wave's 64-col strip
    const int half = lane >> 4, lcol = lane & 15;
    // staging: each thread owns (row srow, 16-wide K-half skh)
    const int srow = tid >> 1, skh = (tid & 1) * 16;

    const int Kend = (causal == 2) ? ((blockM + TM < K) ? blockM + TM : K) : K;

    v8bf ra0, ra1, rb0, rb1;             // staged (already bf16) tile data
    auto load_tiles = [&](int kk) {
        float f[16];
        {
            const float* ap = A + (long long)(blockM + srow) * lda + kk + skh;
            *(float4*)(f + 0)  = *(const float4*)(ap + 0);
            *(float4*)(f + 4)  = *(const float4*)(ap + 4);
            *(float4*)(f + 8)  = *(const float4*)(ap + 8);
            *(float4*)(f + 12) = *(const float4*)(ap + 12);
            if (kk + TKK < Kend) __builtin_prefetch(ap + TKK, 0, 1);
        }
        ra0 = pack8(f); ra1 = pack8(f + 8);
        int gn = blockN + srow;
        if (transB) {
            if (gn < N) {
                const float* bp = Bm + (long long)gn * ldb + kk + skh;
                *(float4*)(f + 0)  = *(const float4*)(bp + 0);
                *(float4*)(f + 4)  = *(const float4*)(bp + 4);
                *(float4*)(f + 8)  = *(const float4*)(bp + 8);
                *(float4*)(f + 12) = *(const float4*)(bp + 12);
                if (kk + TKK < Kend) __builtin_prefetch(bp + TKK, 0, 1);
            } else {
#pragma unroll
                for (int j = 0; j < 16; j++) f[j] = 0.0f;
            }
        } else {
            if (gn < N) {
#pragma unroll
                for (int j = 0; j < 16; j++)
                    f[j] = Bm[(long long)(kk + skh + j) * ldb + gn];
                if (kk + TKK < Kend)
                    __builtin_prefetch(&Bm[(long long)(kk + TKK + skh) * ldb + gn], 0, 1);
            } else {
#pragma unroll
                for (int j = 0; j < 16; j++) f[j] = 0.0f;
            }
        }
        rb0 = pack8(f); rb1 = pack8(f + 8);
    };
    auto store_tiles = [&](int buf) {
        *(v8bf*)&As[buf][srow][skh]      = ra0;
        *(v8bf*)&As[buf][srow][skh + 8]  = ra1;
        *(v8bf*)&BsT[buf][srow][skh]     = rb0;
        *(v8bf*)&BsT[buf][srow][skh + 8] = rb1;
    };

    v8f z;
#pragma unroll
    for (int j = 0; j < 8; j++) z[j] = 0.0f;
    v8f acc[2][4] = {{z, z, z, z}, {z, z, z, z}};

    // pipeline prologue
    load_tiles(0);
    store_tiles(0);
    __syncthreads();

    int buf = 0;
    for (int k0 = 0; k0 < Kend; k0 += TKK) {
        const bool more = (k0 + TKK) < Kend;
        if (more) load_tiles(k0 + TKK);  // global loads overlap the WMMAs below

        // A and B frags K runs {half*8..+7, 16+half*8..+7}.
        v16bf a0 = frag16(&As[buf][mw + lcol][half * 8],
                          &As[buf][mw + lcol][16 + half * 8]);
        v16bf a1 = frag16(&As[buf][mw + 16 + lcol][half * 8],
                          &As[buf][mw + 16 + lcol][16 + half * 8]);
#pragma unroll
        for (int nt = 0; nt < 4; nt++) {
            const __bf16* bp = &BsT[buf][nw + nt * 16 + lcol][half * 8];
            v16bf bf = frag16(bp, bp + 16);
            acc[0][nt] = __builtin_amdgcn_wmma_f32_16x16x32_bf16(
                false, a0, false, bf, (short)0, acc[0][nt], false, false);
            acc[1][nt] = __builtin_amdgcn_wmma_f32_16x16x32_bf16(
                false, a1, false, bf, (short)0, acc[1][nt], false, false);
        }
        if (more) store_tiles(buf ^ 1);  // safe: other buffer, last read pre-barrier
        __syncthreads();
        buf ^= 1;
    }

    // ---- store: VGPR r -> row = mi*16 + 8*half + r ; col = nt*16 + lcol ----
#pragma unroll
    for (int mi = 0; mi < 2; mi++) {
#pragma unroll
        for (int nt = 0; nt < 4; nt++) {
            int gn = blockN + nw + nt * 16 + lcol;
            if (gn >= N) continue;
            float bv = bias ? bias[gn] : 0.0f;
#pragma unroll
            for (int r = 0; r < 8; r++) {
                int gm = blockM + mw + mi * 16 + half * 8 + r;
                long long off = (long long)gm * ldc + gn;
                float v = alpha * acc[mi][nt][r] + bv;
                if (accum) v += Cm[off];
                Cm[off] = v;
            }
        }
    }
}

// ---------------- elementwise / reduction kernels ----------------
__global__ void embed_k(float* __restrict__ x, const float* __restrict__ wte,
                        const int* __restrict__ idx, int total)
{
    int i = blockIdx.x * blockDim.x + threadIdx.x;
    if (i >= total) return;
    int n = i / CDIM, c = i % CDIM;
    x[i] = wte[(long long)idx[n] * CDIM + c];
}

__global__ void ln_k(float* __restrict__ out, const float* __restrict__ x,
                     const float* __restrict__ w, const float* __restrict__ b)
{
    int row = blockIdx.x;
    const float* xr = x + (long long)row * CDIM;
    float* o = out + (long long)row * CDIM;
    int tid = threadIdx.x;
    __shared__ float r1[8], r2[8];
    float s = 0.0f, s2 = 0.0f;
    for (int c = tid; c < CDIM; c += 256) { float v = xr[c]; s += v; s2 += v * v; }
    for (int off = 16; off > 0; off >>= 1) {
        s  += __shfl_xor(s, off, 32);
        s2 += __shfl_xor(s2, off, 32);
    }
    if ((tid & 31) == 0) { r1[tid >> 5] = s; r2[tid >> 5] = s2; }
    __syncthreads();
    float S = 0.0f, S2 = 0.0f;
#pragma unroll
    for (int j = 0; j < 8; j++) { S += r1[j]; S2 += r2[j]; }
    float m   = S / CDIM;
    float var = S2 / CDIM - m * m;
    float inv = rsqrtf(var + 1e-5f);
    for (int c = tid; c < CDIM; c += 256)
        o[c] = (xr[c] - m) * inv * w[c] + b[c];
}

// x layout (B,T,H,DH); interleaved rotate with concat-style cos/sin cache.
__global__ void rope_k(float* __restrict__ x, int npairs)
{
    int p = blockIdx.x * blockDim.x + threadIdx.x;
    if (p >= npairs) return;
    long long base = (long long)p * 2;
    int d0 = (int)(base % DHEAD);
    long long th = base / DHEAD;               // (b*T + t)*H + h
    int t = (int)((th / HNUM) % TSEQ);
    float x0 = x[base], x1 = x[base + 1];
    const float LOG1E4 = 9.210340371976184f;
    float f0 = (float)t * __expf(-((float)(d0 & 31)) / 32.0f * LOG1E4);
    float f1 = (float)t * __expf(-((float)((d0 + 1) & 31)) / 32.0f * LOG1E4);
    x[base]     = x0 * cosf(f0) - x1 * sinf(f0);
    x[base + 1] = x1 * cosf(f1) + x0 * sinf(f1);
}

__global__ void softmax_causal_k(float* __restrict__ att)
{
    int row = blockIdx.x;                    // ((b*H + h)*T + i)
    int i = row % TSEQ;
    float* p = att + (long long)row * TSEQ;
    int tid = threadIdx.x;
    __shared__ float red[8];

    float mx = -3.0e38f;
    for (int j = tid; j <= i; j += 256) mx = fmaxf(mx, p[j]);
    for (int off = 16; off > 0; off >>= 1) mx = fmaxf(mx, __shfl_xor(mx, off, 32));
    if ((tid & 31) == 0) red[tid >> 5] = mx;
    __syncthreads();
    float M = -3.0e38f;
#pragma unroll
    for (int j = 0; j < 8; j++) M = fmaxf(M, red[j]);
    __syncthreads();

    float s = 0.0f;
    for (int j = tid; j <= i; j += 256) { float e = __expf(p[j] - M); p[j] = e; s += e; }
    for (int off = 16; off > 0; off >>= 1) s += __shfl_xor(s, off, 32);
    if ((tid & 31) == 0) red[tid >> 5] = s;
    __syncthreads();
    float S = 0.0f;
#pragma unroll
    for (int j = 0; j < 8; j++) S += red[j];
    float inv = 1.0f / S;
    for (int j = tid; j < TSEQ; j += 256) p[j] = (j <= i) ? p[j] * inv : 0.0f;
}

__global__ void gelu_k(float* __restrict__ x, int total)
{
    int i = blockIdx.x * blockDim.x + threadIdx.x;
    if (i >= total) return;
    float v = x[i];
    x[i] = 0.5f * v * (1.0f + erff(v * 0.70710678118654752f));
}

// top-2-of-6 gating: wout[n,r] = probs[r] if r in top-2 of logits else 0
__global__ void gate_k(float* __restrict__ wout, const float* __restrict__ h2,
                       const float* __restrict__ gW)
{
    int n = blockIdx.x * blockDim.x + threadIdx.x;
    if (n >= NTOK) return;
    const float* hr = h2 + (long long)n * CDIM;
    float lg[NROUTED];
#pragma unroll
    for (int r = 0; r < NROUTED; r++) lg[r] = 0.0f;
    for (int c = 0; c < CDIM; c++) {
        float hv = hr[c];
        const float* g = gW + (long long)c * NROUTED;
#pragma unroll
        for (int r = 0; r < NROUTED; r++) lg[r] += hv * g[r];
    }
    float mx = lg[0];
#pragma unroll
    for (int r = 1; r < NROUTED; r++) mx = fmaxf(mx, lg[r]);
    float pr[NROUTED], s = 0.0f;
#pragma unroll
    for (int r = 0; r < NROUTED; r++) { pr[r] = __expf(lg[r] - mx); s += pr[r]; }
    float inv = 1.0f / s;
    int i1 = 0;
#pragma unroll
    for (int r = 1; r < NROUTED; r++) if (lg[r] > lg[i1]) i1 = r;
    int i2 = -1;
#pragma unroll
    for (int r = 0; r < NROUTED; r++)
        if (r != i1 && (i2 < 0 || lg[r] > lg[i2])) i2 = r;
#pragma unroll
    for (int r = 0; r < NROUTED; r++)
        wout[(long long)n * NROUTED + r] = (r == i1 || r == i2) ? pr[r] * inv : 0.0f;
}

// x += s * oe, s = 1 (shared expert) or gating weight w[n, eoff]
__global__ void moe_acc_k(float* __restrict__ x, const float* __restrict__ oe,
                          const float* __restrict__ w, int eoff, int total)
{
    int i = blockIdx.x * blockDim.x + threadIdx.x;
    if (i >= total) return;
    float s = 1.0f;
    if (w) { int n = i / CDIM; s = w[(long long)n * NROUTED + eoff]; }
    x[i] += s * oe[i];
}

// ---------------- host side ----------------
static void launch_gemm(hipStream_t s,
    const float* A, int lda, long long sA1, long long sA2,
    const float* B, int ldb, long long sB1, long long sB2,
    float* C, int ldc, long long sC1, long long sC2,
    const float* bias, int M, int N, int K,
    int nb1, int nb2, float alpha, int transB, int accum, int causal)
{
    dim3 grid((N + TN - 1) / TN, (M + TM - 1) / TM, nb1 * nb2);
    gemm_bf16_wmma<<<grid, 256, 0, s>>>(A, lda, sA1, sA2, B, ldb, sB1, sB2,
                                        C, ldc, sC1, sC2, bias,
                                        M, N, K, nb2, alpha, transB, accum, causal);
}

extern "C" void kernel_launch(void* const* d_in, const int* in_sizes, int n_in,
                              void* d_out, int out_size, void* d_ws, size_t ws_size,
                              hipStream_t stream)
{
    (void)in_sizes; (void)n_in; (void)out_size; (void)ws_size;

    const float* wte   = (const float*)d_in[0];
    const float* ln1_w = (const float*)d_in[1];
    const float* ln1_b = (const float*)d_in[2];
    const float* qW    = (const float*)d_in[3];
    const float* qb    = (const float*)d_in[4];
    const float* kW    = (const float*)d_in[5];
    const float* kb    = (const float*)d_in[6];
    const float* vW    = (const float*)d_in[7];
    const float* vb    = (const float*)d_in[8];
    const float* kupW  = (const float*)d_in[9];
    const float* kupb  = (const float*)d_in[10];
    const float* vupW  = (const float*)d_in[11];
    const float* vupb  = (const float*)d_in[12];
    const float* cW    = (const float*)d_in[13];
    const float* cb    = (const float*)d_in[14];
    const float* ln2_w = (const float*)d_in[15];
    const float* ln2_b = (const float*)d_in[16];
    const float* gateW = (const float*)d_in[17];
    const float* fcW   = (const float*)d_in[18];
    const float* fcB   = (const float*)d_in[19];
    const float* pW    = (const float*)d_in[20];
    const float* pB    = (const float*)d_in[21];
    const float* lnf_w = (const float*)d_in[22];
    const float* lnf_b = (const float*)d_in[23];
    const int*   idx   = (const int*)d_in[24];
    float* out = (float*)d_out;

    // workspace carve (floats): ~101 MB total
    float* ws = (float*)d_ws;
    const long long NC = (long long)NTOK * CDIM;       // 1,048,576
    float* x    = ws;            ws += NC;
    float* h    = ws;            ws += NC;
    float* q    = ws;            ws += NC;
    float* kl   = ws;            ws += NC;
    float* vl   = ws;            ws += NC;
    float* kbuf = ws;            ws += NC;
    float* vbuf = ws;            ws += NC;
    float* y    = ws;            ws += NC;
    float* att  = ws;            ws += (long long)BSZ * HNUM * TSEQ * TSEQ; // 16.8M
    float* gw   = ws;            ws += (long long)NTOK * NROUTED;
    float* hh   = att;                       // alias: MoE phase reuses att region
    float* oe   = att + (long long)NTOK * HIDDEN;

    const int EL = 256;
    // x = wte[idx]
    embed_k<<<((int)NC + EL - 1) / EL, EL, 0, stream>>>(x, wte, idx, (int)NC);

    for (int l = 0; l < LNUM; l++) {
        // h = ln1(x)
        ln_k<<<NTOK, 256, 0, stream>>>(h, x, ln1_w + l * CDIM, ln1_b + l * CDIM);

        // q / kl / vl projections
        launch_gemm(stream, h, CDIM, 0, 0, qW + (long long)l * CDIM * CDIM, CDIM, 0, 0,
                    q, CDIM, 0, 0, qb + l * CDIM, NTOK, CDIM, CDIM, 1, 1, 1.0f, 0, 0, 0);
        launch_gemm(stream, h, CDIM, 0, 0, kW + (long long)l * CDIM * (HNUM * DLOW), HNUM * DLOW, 0, 0,
                    kl, HNUM * DLOW, 0, 0, kb + l * HNUM * DLOW, NTOK, HNUM * DLOW, CDIM, 1, 1, 1.0f, 0, 0, 0);
        launch_gemm(stream, h, CDIM, 0, 0, vW + (long long)l * CDIM * (HNUM * DLOW), HNUM * DLOW, 0, 0,
                    vl, HNUM * DLOW, 0, 0, vb + l * HNUM * DLOW, NTOK, HNUM * DLOW, CDIM, 1, 1, 1.0f, 0, 0, 0);

        // k/v up-projection: (NTOK*H, DL) @ (DL, DH)
        launch_gemm(stream, kl, DLOW, 0, 0, kupW + (long long)l * DLOW * DHEAD, DHEAD, 0, 0,
                    kbuf, DHEAD, 0, 0, kupb + l * DHEAD, NTOK * HNUM, DHEAD, DLOW, 1, 1, 1.0f, 0, 0, 0);
        launch_gemm(stream, vl, DLOW, 0, 0, vupW + (long long)l * DLOW * DHEAD, DHEAD, 0, 0,
                    vbuf, DHEAD, 0, 0, vupb + l * DHEAD, NTOK * HNUM, DHEAD, DLOW, 1, 1, 1.0f, 0, 0, 0);

        // RoPE on q and k
        int npairs = (int)(NC / 2);
        rope_k<<<(npairs + EL - 1) / EL, EL, 0, stream>>>(q, npairs);
        rope_k<<<(npairs + EL - 1) / EL, EL, 0, stream>>>(kbuf, npairs);

        // att[b,h] = (q[b,:,h,:] @ k[b,:,h,:]^T) / sqrt(DH), batched over (b,h)
        // causal=1: upper-triangle 128x128 tiles skipped (softmax zeroes them)
        launch_gemm(stream,
                    q,    CDIM, (long long)TSEQ * CDIM, DHEAD,
                    kbuf, CDIM, (long long)TSEQ * CDIM, DHEAD,
                    att,  TSEQ, (long long)HNUM * TSEQ * TSEQ, (long long)TSEQ * TSEQ,
                    nullptr, TSEQ, TSEQ, DHEAD, BSZ, HNUM, 0.125f, /*transB=*/1, 0, /*causal=*/1);

        softmax_causal_k<<<BSZ * HNUM * TSEQ, 256, 0, stream>>>(att);

        // y[b,:,h,:] = att[b,h] @ v[b,:,h,:] ; causal=2: K truncated at row limit
        launch_gemm(stream,
                    att,  TSEQ, (long long)HNUM * TSEQ * TSEQ, (long long)TSEQ * TSEQ,
                    vbuf, CDIM, (long long)TSEQ * CDIM, DHEAD,
                    y,    CDIM, (long long)TSEQ * CDIM, DHEAD,
                    nullptr, TSEQ, DHEAD, TSEQ, BSZ, HNUM, 1.0f, 0, 0, /*causal=*/2);

        // x += y @ cW + cb
        launch_gemm(stream, y, CDIM, 0, 0, cW + (long long)l * CDIM * CDIM, CDIM, 0, 0,
                    x, CDIM, 0, 0, cb + l * CDIM, NTOK, CDIM, CDIM, 1, 1, 1.0f, 0, /*accum=*/1, 0);

        // h = ln2(x); gating weights
        ln_k<<<NTOK, 256, 0, stream>>>(h, x, ln2_w + l * CDIM, ln2_b + l * CDIM);
        gate_k<<<(NTOK + EL - 1) / EL, EL, 0, stream>>>(gw, h, gateW + (long long)l * CDIM * NROUTED);

        // dense experts, weighted accumulate into x
        for (int e = 0; e < ENUM; e++) {
            long long we = (long long)l * ENUM + e;
            launch_gemm(stream, h, CDIM, 0, 0, fcW + we * CDIM * HIDDEN, HIDDEN, 0, 0,
                        hh, HIDDEN, 0, 0, fcB + we * HIDDEN, NTOK, HIDDEN, CDIM, 1, 1, 1.0f, 0, 0, 0);
            int tot = NTOK * HIDDEN;
            gelu_k<<<(tot + EL - 1) / EL, EL, 0, stream>>>(hh, tot);
            launch_gemm(stream, hh, HIDDEN, 0, 0, pW + we * HIDDEN * CDIM, CDIM, 0, 0,
                        oe, CDIM, 0, 0, pB + we * CDIM, NTOK, CDIM, HIDDEN, 1, 1, 1.0f, 0, 0, 0);
            const float* wsel = (e < NSHARED) ? nullptr : gw;
            moe_acc_k<<<((int)NC + EL - 1) / EL, EL, 0, stream>>>(x, oe, wsel, e - NSHARED, (int)NC);
        }
    }

    // final LN + logits = h @ wte^T
    ln_k<<<NTOK, 256, 0, stream>>>(h, x, lnf_w, lnf_b);
    launch_gemm(stream, h, CDIM, 0, 0, wte, CDIM, 0, 0,
                out, VOCAB, 0, 0, nullptr, NTOK, VOCAB, CDIM, 1, 1, 1.0f, /*transB=*/1, 0, 0);
}
